// ConcatAttention_Param_80255758893461
// MI455X (gfx1250) — compile-verified, matrix operand order checked by reading, not run on hardware
//
#include <hip/hip_runtime.h>
#include <hip/hip_bf16.h>

// CDNA5 / gfx1250: wave32, WMMA 16x16x32 bf16 with f32 accumulate.
typedef __attribute__((ext_vector_type(16))) __bf16 v16bf;
typedef __attribute__((ext_vector_type(8)))  float  v8f;

#define INPUT_SIZE 1024
#define HIDDEN     1024
#define BATCH      32
#define SEQ        2048
#define ROWS       (BATCH * SEQ)

#define LDS_STRIDE 1032   // bf16 elems per row in LDS (pad 8 -> 2064B stride, bank-conflict free b128)
#define BLOCK_ROWS 64     // 4 M-tiles of 16

// ---------- helpers ----------
union FragU { uint4 q[2]; v16bf v; };

__device__ __forceinline__ v16bf load_frag(const __bf16* p) {
  // ISA 16-bit A/B fragment: lane holds K[base .. base+7] and K[base+16 .. base+23]
  // (caller already applied the sel = lane>=16 ? +8 : 0 shift to p).
  FragU u;
  u.q[0] = *reinterpret_cast<const uint4*>(p);
  u.q[1] = *reinterpret_cast<const uint4*>(p + 16);
  return u.v;
}

__device__ __forceinline__ v8f vzero8() {
  v8f z = {0.f, 0.f, 0.f, 0.f, 0.f, 0.f, 0.f, 0.f};
  return z;
}

// ---------- W split: fp32 -> bf16 hi + bf16 lo ----------
__global__ __launch_bounds__(256)
void convert_w_kernel(const float* __restrict__ ln_w,
                      __bf16* __restrict__ Whi, __bf16* __restrict__ Wlo) {
  int base = (blockIdx.x * 256 + threadIdx.x) * 4;
#pragma unroll
  for (int t = 0; t < 4; ++t) {
    int idx = base + t;                      // flat index into W_h [1024][1024]
    int j = idx >> 10, k = idx & 1023;
    float v = ln_w[(size_t)j * 2048 + k];    // ln_w row has 2048 cols; W_h = [:, :1024]
    __bf16 hi = (__bf16)v;
    Whi[idx] = hi;
    Wlo[idx] = (__bf16)(v - (float)hi);
  }
}

// ---------- bias = ln_b + W_vq @ vq (one wave per output row) ----------
__global__ __launch_bounds__(256)
void bias_kernel(const float* __restrict__ ln_w, const float* __restrict__ ln_b,
                 const float* __restrict__ vq, float* __restrict__ bias) {
  const int wave = threadIdx.x >> 5, lane = threadIdx.x & 31;
  const int j = blockIdx.x * 8 + wave;
  const float* wr = ln_w + (size_t)j * 2048 + 1024;   // W_vq row j
  float s = 0.f;
  for (int k = lane; k < 1024; k += 32) s += wr[k] * vq[k];
  s += __shfl_xor(s, 16, 32);
  s += __shfl_xor(s,  8, 32);
  s += __shfl_xor(s,  4, 32);
  s += __shfl_xor(s,  2, 32);
  s += __shfl_xor(s,  1, 32);
  if (lane == 0) bias[j] = ln_b[j] + s;
}

// ---------- fused GEMM + tanh + dot(v_w): s[row] = sum_j tanh(h@Wh^T + bias)_j * v_w[j] ----------
__global__ __launch_bounds__(256, 1)
void fused_score_kernel(const float* __restrict__ h,
                        const __bf16* __restrict__ Whi,
                        const __bf16* __restrict__ Wlo,
                        const float* __restrict__ bias,
                        const float* __restrict__ vw,
                        float* __restrict__ s_out) {
  extern __shared__ char smem[];
  __bf16* Ahi = (__bf16*)smem;                            // [64][1032] bf16 hi
  __bf16* Alo = Ahi + BLOCK_ROWS * LDS_STRIDE;            // [64][1032] bf16 lo
  float*  sbuf = (float*)(Alo + BLOCK_ROWS * LDS_STRIDE); // [8 waves][64 rows]

  const int tid  = threadIdx.x;
  const int wave = tid >> 5;
  const int lane = tid & 31;
  const int ln   = lane & 15;
  const int sel  = lane >> 4;

  const long rowbase = (long)blockIdx.x * BLOCK_ROWS;

  // --- load 64x1024 fp32 tile of h, split to bf16 hi/lo in LDS ---
  const float4* h4 = reinterpret_cast<const float4*>(h + rowbase * 1024);
  for (int e = tid; e < BLOCK_ROWS * 1024 / 4; e += 256) {
    int row = e >> 8;              // 256 float4 per row
    int k   = (e & 255) * 4;
    float4 v = h4[e];
    __bf16 h0 = (__bf16)v.x, h1 = (__bf16)v.y, h2 = (__bf16)v.z, h3 = (__bf16)v.w;
    union { __bf16 b[4]; uint2 u; } ph, pl;
    ph.b[0] = h0; ph.b[1] = h1; ph.b[2] = h2; ph.b[3] = h3;
    pl.b[0] = (__bf16)(v.x - (float)h0);
    pl.b[1] = (__bf16)(v.y - (float)h1);
    pl.b[2] = (__bf16)(v.z - (float)h2);
    pl.b[3] = (__bf16)(v.w - (float)h3);
    *reinterpret_cast<uint2*>(Ahi + row * LDS_STRIDE + k) = ph.u;
    *reinterpret_cast<uint2*>(Alo + row * LDS_STRIDE + k) = pl.u;
  }
  __syncthreads();

  v8f sacc[4];
#pragma unroll
  for (int m = 0; m < 4; ++m) sacc[m] = vzero8();

  const __bf16* arow = Ahi + (size_t)ln * LDS_STRIDE + sel * 8;
  const __bf16* arol = Alo + (size_t)ln * LDS_STRIDE + sel * 8;

  for (int p = 0; p < 2; ++p) {              // this wave's 8 N-tiles, in groups of 4
    const int ntb = wave * 8 + p * 4;

    const __bf16* bhp[4];
    const __bf16* blp[4];
#pragma unroll
    for (int nt = 0; nt < 4; ++nt) {
      bhp[nt] = Whi + (size_t)((ntb + nt) * 16 + ln) * 1024 + sel * 8;
      blp[nt] = Wlo + (size_t)((ntb + nt) * 16 + ln) * 1024 + sel * 8;
    }

    v8f acc[4][4];                           // [nt][m] -> 128 VGPRs
#pragma unroll
    for (int nt = 0; nt < 4; ++nt)
#pragma unroll
      for (int m = 0; m < 4; ++m) acc[nt][m] = vzero8();

    for (int ks = 0; ks < 32; ++ks) {        // K = 1024 in steps of 32
      const int kb = ks * 32;
      v16bf bh[4], bl[4];
#pragma unroll
      for (int nt = 0; nt < 4; ++nt) {
        bh[nt] = load_frag(bhp[nt] + kb);
        bl[nt] = load_frag(blp[nt] + kb);
      }
#pragma unroll
      for (int m = 0; m < 4; ++m) {
        v16bf ah = load_frag(arow + m * 16 * LDS_STRIDE + kb);
        v16bf al = load_frag(arol + m * 16 * LDS_STRIDE + kb);
        // fp32-accurate product via bf16 split: Ah*Bh + Ah*Bl + Al*Bh
#pragma unroll
        for (int nt = 0; nt < 4; ++nt) {
          acc[nt][m] = __builtin_amdgcn_wmma_f32_16x16x32_bf16(false, ah, false, bh[nt], (short)0, acc[nt][m], false, false);
          acc[nt][m] = __builtin_amdgcn_wmma_f32_16x16x32_bf16(false, ah, false, bl[nt], (short)0, acc[nt][m], false, false);
          acc[nt][m] = __builtin_amdgcn_wmma_f32_16x16x32_bf16(false, al, false, bh[nt], (short)0, acc[nt][m], false, false);
        }
      }
    }

    // epilogue: C/D layout -> lane holds column j = nt*16+ln, rows m*16 + sel*8 + g
#pragma unroll
    for (int nt = 0; nt < 4; ++nt) {
      const int j = (ntb + nt) * 16 + ln;
      const float bj = bias[j], wj = vw[j];
#pragma unroll
      for (int m = 0; m < 4; ++m) {
#pragma unroll
        for (int g = 0; g < 8; ++g) {
          sacc[m][g] += tanhf(acc[nt][m][g] + bj) * wj;
        }
      }
    }
  }

  // reduce over the 16 j-lanes of each half-wave, then across waves via LDS
#pragma unroll
  for (int m = 0; m < 4; ++m) {
#pragma unroll
    for (int g = 0; g < 8; ++g) {
      float v = sacc[m][g];
      v += __shfl_xor(v, 1, 32);
      v += __shfl_xor(v, 2, 32);
      v += __shfl_xor(v, 4, 32);
      v += __shfl_xor(v, 8, 32);
      if (ln == 0) sbuf[wave * 64 + m * 16 + sel * 8 + g] = v;
    }
  }
  __syncthreads();
  if (tid < BLOCK_ROWS) {
    float t = 0.f;
#pragma unroll
    for (int w = 0; w < 8; ++w) t += sbuf[w * 64 + tid];
    s_out[rowbase + tid] = t;
  }
}

// ---------- masked softmax over L per batch ----------
__global__ __launch_bounds__(256)
void softmax_kernel(const float* __restrict__ s, const int* __restrict__ mask,
                    float* __restrict__ a) {
  __shared__ float red[256];
  const int b = blockIdx.x, tid = threadIdx.x;
  const float* sb = s + b * SEQ;
  const int*   mb = mask + b * SEQ;

  float mx = -3.0e38f;
  for (int e = tid; e < SEQ; e += 256) {
    float v = sb[e] - (mb[e] ? 0.f : 10000.f);
    mx = fmaxf(mx, v);
  }
  red[tid] = mx; __syncthreads();
  for (int o = 128; o > 0; o >>= 1) { if (tid < o) red[tid] = fmaxf(red[tid], red[tid + o]); __syncthreads(); }
  mx = red[0]; __syncthreads();

  float sum = 0.f;
  for (int e = tid; e < SEQ; e += 256) {
    float v = sb[e] - (mb[e] ? 0.f : 10000.f);
    sum += __expf(v - mx);
  }
  red[tid] = sum; __syncthreads();
  for (int o = 128; o > 0; o >>= 1) { if (tid < o) red[tid] += red[tid + o]; __syncthreads(); }
  const float inv = 1.f / red[0];

  for (int e = tid; e < SEQ; e += 256) {
    float v = sb[e] - (mb[e] ? 0.f : 10000.f);
    a[b * SEQ + e] = __expf(v - mx) * inv;
  }
}

// ---------- r partials: part[ls][b][i] = sum_{l in chunk ls} a[b,l] * h[b,l,i] ----------
__global__ __launch_bounds__(256)
void weighted_partial_kernel(const float* __restrict__ h, const float* __restrict__ a,
                             float* __restrict__ part) {
  __shared__ float aw[128];
  const int b  = blockIdx.x >> 4;
  const int ls = blockIdx.x & 15;
  const int tid = threadIdx.x;
  const int l0 = ls * 128;
  if (tid < 128) aw[tid] = a[b * SEQ + l0 + tid];
  __syncthreads();

  float4 acc = make_float4(0.f, 0.f, 0.f, 0.f);
  const float4* hb = reinterpret_cast<const float4*>(h + ((size_t)b * SEQ + l0) * 1024);
  for (int l = 0; l < 128; ++l) {
    float w = aw[l];
    float4 hv = hb[l * 256 + tid];
    acc.x += w * hv.x; acc.y += w * hv.y; acc.z += w * hv.z; acc.w += w * hv.w;
  }
  *reinterpret_cast<float4*>(part + ((size_t)(ls * 32 + b)) * 1024 + tid * 4) = acc;
}

// ---------- final reduce over the 16 L-chunks ----------
__global__ __launch_bounds__(256)
void final_reduce_kernel(const float* __restrict__ part, float* __restrict__ out) {
  int idx = blockIdx.x * 256 + threadIdx.x;   // 0 .. 32*1024-1
  int b = idx >> 10, i = idx & 1023;
  float t = 0.f;
#pragma unroll
  for (int ls = 0; ls < 16; ++ls) t += part[((size_t)(ls * 32 + b)) * 1024 + i];
  out[idx] = t;
}

// ---------- launch ----------
extern "C" void kernel_launch(void* const* d_in, const int* in_sizes, int n_in,
                              void* d_out, int out_size, void* d_ws, size_t ws_size,
                              hipStream_t stream) {
  const float* h    = (const float*)d_in[0];
  const int*   mask = (const int*)d_in[1];     // bool mask passed as int32 per harness dtype rules
  const float* ln_w = (const float*)d_in[2];
  const float* ln_b = (const float*)d_in[3];
  const float* v_w  = (const float*)d_in[4];
  const float* vq   = (const float*)d_in[5];
  float* out = (float*)d_out;

  // workspace layout (~6.5 MiB total)
  char* ws = (char*)d_ws;
  __bf16* Whi  = (__bf16*)ws;                                    // 2 MiB
  __bf16* Wlo  = (__bf16*)(ws + (2u << 20));                     // 2 MiB
  float*  bias = (float*) (ws + (4u << 20));                     // 4 KiB
  float*  s    = (float*) (ws + (4u << 20) + 4096);              // 256 KiB
  float*  a    = (float*) (ws + (4u << 20) + 4096 + (256u<<10)); // 256 KiB
  float*  part = (float*) (ws + (4u << 20) + 4096 + (512u<<10)); // 2 MiB

  convert_w_kernel<<<1024, 256, 0, stream>>>(ln_w, Whi, Wlo);
  bias_kernel<<<128, 256, 0, stream>>>(ln_w, ln_b, vq, bias);

  size_t smem = (size_t)2 * BLOCK_ROWS * LDS_STRIDE * sizeof(__bf16)
              + 8 * BLOCK_ROWS * sizeof(float);                  // 266,240 B < 320 KB WGP LDS
  fused_score_kernel<<<ROWS / BLOCK_ROWS, 256, smem, stream>>>(h, Whi, Wlo, bias, v_w, s);

  softmax_kernel<<<BATCH, 256, 0, stream>>>(s, mask, a);
  weighted_partial_kernel<<<BATCH * 16, 256, 0, stream>>>(h, a, part);
  final_reduce_kernel<<<(BATCH * 1024) / 256, 256, 0, stream>>>(part, out);
}
